// MixerDiffAttention_8770323219187
// MI455X (gfx1250) — compile-verified, matrix operand order checked by reading, not run on hardware
//
#include <hip/hip_runtime.h>
#include <hip/hip_bf16.h>

// ---------- CDNA5 WMMA types ----------
typedef __attribute__((ext_vector_type(16))) __bf16 v16bf;
typedef __attribute__((ext_vector_type(8)))  float  v8f;
typedef unsigned short u16;
typedef unsigned int   u32;
typedef unsigned int   u32x4 __attribute__((ext_vector_type(4)));
typedef int            i32x4 __attribute__((ext_vector_type(4)));
typedef int            i32x8 __attribute__((ext_vector_type(8)));

union FragU { v16bf v; uint4 q[2]; };

// Native bf16 conversion (backend emits v_cvt_*bf16* on gfx1250).
static __device__ __forceinline__ u16 f2bf(float a) {
  __bf16 h = (__bf16)a;
  return __builtin_bit_cast(u16, h);
}
static __device__ __forceinline__ u32 pk2bf(float a, float b) {
  return (u32)f2bf(a) | ((u32)f2bf(b) << 16);
}
// Load two 16B slices (8 bf16 each) from LDS into a v16bf fragment.
static __device__ __forceinline__ v16bf ldsFrag(const u16* p, int off1, int off2) {
  FragU f;
  f.q[0] = *(const uint4*)(p + off1);
  f.q[1] = *(const uint4*)(p + off2);
  return f.v;
}
static __device__ __forceinline__ v8f wmma_bf16(v16bf a, v16bf b, v8f c) {
  return __builtin_amdgcn_wmma_f32_16x16x32_bf16(false, a, false, b, (short)0, c,
                                                 false, false);
}
static __device__ __forceinline__ float fast_rcp(float x) {
  return __builtin_amdgcn_rcpf(x);
}

// DPP butterfly over 16-lane halves (VALU only; no LDS round-trips).
template <int CTRL>
static __device__ __forceinline__ float dppf(float x) {
  int xi = __builtin_bit_cast(int, x);
  return __builtin_bit_cast(float,
      __builtin_amdgcn_update_dpp(xi, xi, CTRL, 0xf, 0xf, true));
}
static __device__ __forceinline__ float red16_max(float x) {
  x = fmaxf(x, dppf<0xB1>(x));   // quad_perm [1,0,3,2]  (xor 1)
  x = fmaxf(x, dppf<0x4E>(x));   // quad_perm [2,3,0,1]  (xor 2)
  x = fmaxf(x, dppf<0x141>(x));  // row_half_mirror      (xor 7)
  x = fmaxf(x, dppf<0x140>(x));  // row_mirror           (xor 15)
  return x;
}
static __device__ __forceinline__ float red16_sum(float x) {
  x += dppf<0xB1>(x);
  x += dppf<0x4E>(x);
  x += dppf<0x141>(x);
  x += dppf<0x140>(x);
  return x;
}

// ---------------------------------------------------------------------
// Tensor Data Mover: 2D bf16 tile load Global -> LDS (ISA ch.7/8).
// D# group0: count=1 | lds_addr | global_addr[56:0] | type=2
// D# group1: data_size=2B, tensor_dim0/1, tile_dim0/1, dim0_stride
// Groups 2/3 zero (<=2D tensor). Issued once per block by wave 0.
// ---------------------------------------------------------------------
static __device__ __forceinline__ void tdm_load_2d_bf16(
    const void* gaddr, u32 lds_off, u32 tile_x, u32 tile_y, u32 stride0) {
  unsigned long long ga = (unsigned long long)gaddr;
  u32x4 g0;
  g0[0] = 1u;                                            // count=1 (user D#)
  g0[1] = lds_off;                                       // lds_addr
  g0[2] = (u32)ga;                                       // global_addr[31:0]
  g0[3] = (u32)((ga >> 32) & 0x1FFFFFFu) | (2u << 30);   // addr[56:32], type=2
  i32x8 g1;
  g1[0] = 1 << 16;                                       // data_size=1 -> 2 bytes
  g1[1] = (int)((tile_x & 0xFFFFu) << 16);               // tensor_dim0[15:0]
  g1[2] = (int)((tile_x >> 16) | ((tile_y & 0xFFFFu) << 16));  // dim0 hi | dim1 lo
  g1[3] = (int)((tile_y >> 16) | ((tile_x & 0xFFFFu) << 16));  // dim1 hi | tile_dim0
  g1[4] = (int)(tile_y & 0xFFFFu);                       // tile_dim1 (tile_dim2=0)
  g1[5] = (int)stride0;                                  // tensor_dim0_stride lo
  g1[6] = 0;                                             // stride0 hi | dim1_stride lo
  g1[7] = 0;
  i32x4 z4 = {0, 0, 0, 0};
#if defined(__clang_major__) && (__clang_major__ >= 23)
  i32x8 z8 = {0, 0, 0, 0, 0, 0, 0, 0};
  __builtin_amdgcn_tensor_load_to_lds(g0, g1, z4, z4, z8, 0);
#else
  __builtin_amdgcn_tensor_load_to_lds(g0, g1, z4, z4, 0);
#endif
}

// ---------- problem constants ----------
constexpr int   Bc   = 2;
constexpr int   Tc   = 2048;
constexpr int   DM   = 2048;
constexpr int   NH   = 16;
constexpr int   HD   = 128;
constexpr int   HPc  = 8;
constexpr int   QKVN = 6144;          // 16 heads * 384 (q|k|v interleaved per head)
constexpr float LAMBDA_INIT = 0.7008206670670481f;
constexpr float EPSc = 1e-6f;

// =====================================================================
// Kernel 1: bf16 WMMA GEMM  C[M,N] = A[M,K] * W[K,N]   (compile-time SiLU)
// Block: 256 thr (8 waves), tile 128x128, K-step 32, wave tile 32x64.
// =====================================================================
template <int DO_SILU>
__global__ __launch_bounds__(256) void gemm_bf16_kernel(
    const float* __restrict__ A, const float* __restrict__ W,
    float* __restrict__ C, int M, int N, int K) {
  __shared__ alignas(16) u16 As[128 * 32];   // [m][k] bf16
  __shared__ alignas(16) u16 Bs[128 * 32];   // [n][k] bf16 (transposed)

  const int tid  = threadIdx.x;
  const int lane = tid & 31, wid = tid >> 5;
  const int half = lane >> 4, l15 = lane & 15;
  const int mo = (wid >> 1) * 32;
  const int no = (wid & 1) * 64;
  const int Mb = blockIdx.y * 128, Nb = blockIdx.x * 128;
  const int ko1 = 8 * half, ko2 = 16 + 8 * half;

  v8f acc[2][4];
#pragma unroll
  for (int i = 0; i < 2; i++)
#pragma unroll
    for (int j = 0; j < 4; j++)
#pragma unroll
      for (int e = 0; e < 8; e++) acc[i][j][e] = 0.f;

  for (int k0 = 0; k0 < K; k0 += 32) {
    __syncthreads();
#pragma unroll
    for (int i = 0; i < 4; i++) {
      int f = tid + i * 256;
      int row = f >> 3, c4 = f & 7;
      float4 v = *(const float4*)(A + (size_t)(Mb + row) * K + k0 + c4 * 4);
      uint2 p; p.x = pk2bf(v.x, v.y); p.y = pk2bf(v.z, v.w);
      *(uint2*)(&As[row * 32 + c4 * 4]) = p;
    }
#pragma unroll
    for (int i = 0; i < 4; i++) {
      int f = tid + i * 256;
      int row = f >> 5, c4 = f & 31;
      float4 v = *(const float4*)(W + (size_t)(k0 + row) * N + Nb + c4 * 4);
      int n0 = c4 * 4;
      Bs[(n0 + 0) * 32 + row] = f2bf(v.x);
      Bs[(n0 + 1) * 32 + row] = f2bf(v.y);
      Bs[(n0 + 2) * 32 + row] = f2bf(v.z);
      Bs[(n0 + 3) * 32 + row] = f2bf(v.w);
    }
    if (k0 + 32 < K) {
      __builtin_prefetch(A + (size_t)(Mb + (tid >> 1)) * K + (k0 + 32) + (tid & 1) * 16,
                         0, 0);
      __builtin_prefetch(W + (size_t)(k0 + 32 + (tid >> 3)) * N + Nb + (tid & 7) * 16,
                         0, 0);
    }
    __syncthreads();

    v16bf a0 = ldsFrag(As + (mo +  0 + l15) * 32, ko1, ko2);
    v16bf a1 = ldsFrag(As + (mo + 16 + l15) * 32, ko1, ko2);
    v16bf bf0 = ldsFrag(Bs + (no +  0 + l15) * 32, ko1, ko2);
    v16bf bf1 = ldsFrag(Bs + (no + 16 + l15) * 32, ko1, ko2);
    v16bf bf2 = ldsFrag(Bs + (no + 32 + l15) * 32, ko1, ko2);
    v16bf bf3 = ldsFrag(Bs + (no + 48 + l15) * 32, ko1, ko2);
    acc[0][0] = wmma_bf16(a0, bf0, acc[0][0]);
    acc[0][1] = wmma_bf16(a0, bf1, acc[0][1]);
    acc[0][2] = wmma_bf16(a0, bf2, acc[0][2]);
    acc[0][3] = wmma_bf16(a0, bf3, acc[0][3]);
    acc[1][0] = wmma_bf16(a1, bf0, acc[1][0]);
    acc[1][1] = wmma_bf16(a1, bf1, acc[1][1]);
    acc[1][2] = wmma_bf16(a1, bf2, acc[1][2]);
    acc[1][3] = wmma_bf16(a1, bf3, acc[1][3]);
  }

#pragma unroll
  for (int i = 0; i < 2; i++)
#pragma unroll
    for (int j = 0; j < 4; j++)
#pragma unroll
      for (int e = 0; e < 8; e++) {
        int row = Mb + mo + 16 * i + e + 8 * half;
        int col = Nb + no + 16 * j + l15;
        float v = acc[i][j][e];
        if (DO_SILU) v = v * fast_rcp(1.f + __expf(-v));
        C[(size_t)row * N + col] = v;
      }
}

// =====================================================================
// Kernel 2: per-(b,t,head) RMSNorm + RoPE -> bf16 q̂/k̂, branch-split.
// =====================================================================
__global__ __launch_bounds__(128) void prep_qk_kernel(
    const float* __restrict__ qkv, u16* __restrict__ qh, u16* __restrict__ kh) {
  __shared__ float sq[128], sk[128];
  const int d = threadIdx.x;
  const int lin = blockIdx.x;
  const int h = lin & 15;
  const int t = (lin >> 4) & (Tc - 1);
  const int b = lin >> 15;

  const size_t base = (size_t)(b * Tc + t) * QKVN + (size_t)h * 384;
  const float qv = qkv[base + d];
  const float kv = qkv[base + 128 + d];

  sq[d] = qv * qv; sk[d] = kv * kv;
  __syncthreads();
  for (int s = 64; s > 0; s >>= 1) {
    if (d < s) { sq[d] += sq[d + s]; sk[d] += sk[d + s]; }
    __syncthreads();
  }
  const float rq = rsqrtf(sq[0] * (1.f / 128.f) + EPSc);
  const float rk = rsqrtf(sk[0] * (1.f / 128.f) + EPSc);
  __syncthreads();
  sq[d] = qv * rq; sk[d] = kv * rk;
  __syncthreads();

  const int j = d & 63;
  const float inv = __powf(10000.f, -(float)(2 * j) * (1.f / 128.f));
  const float ang = (float)t * inv;
  const float c = __cosf(ang), s = __sinf(ang);
  float qo, ko;
  if (d < 64) {
    qo =  sq[d] * c + sq[d + 64] * s;
    ko =  sk[d] * c + sk[d + 64] * s;
  } else {
    qo = -sq[d - 64] * s + sq[d] * c;
    ko = -sk[d - 64] * s + sk[d] * c;
  }
  const int br = h & 1, hp = h >> 1;
  const size_t oi = (((size_t)(b * 2 + br) * HPc + hp) * Tc + t) * HD + d;
  qh[oi] = f2bf(qo);
  kh[oi] = f2bf(ko);
}

// =====================================================================
// Kernel 2b: V transpose+convert: qkv f32 -> vT[b*HP+hp][col 256][t] bf16.
// =====================================================================
__global__ __launch_bounds__(256) void vtrans_kernel(
    const float* __restrict__ qkv, u16* __restrict__ vT) {
  __shared__ alignas(16) u16 tile[64][72];  // pad 8 u16
  const int tid = threadIdx.x;
  const int t0 = blockIdx.x * 64;
  const int c0 = blockIdx.y * 64;
  const int bh = blockIdx.z;            // b*HP + hp
  const int b = bh >> 3, hp = bh & 7;
  const int hh = 2 * hp + (c0 >> 7);
  const int cin = 256 + (c0 & 127);

#pragma unroll
  for (int i = 0; i < 4; i++) {
    int f = tid + i * 256;
    int tl = f >> 4, c4 = f & 15;
    float4 v = *(const float4*)(qkv + (size_t)(b * Tc + t0 + tl) * QKVN +
                                (size_t)hh * 384 + cin + c4 * 4);
    int cl = c4 * 4;
    tile[cl + 0][tl] = f2bf(v.x);
    tile[cl + 1][tl] = f2bf(v.y);
    tile[cl + 2][tl] = f2bf(v.z);
    tile[cl + 3][tl] = f2bf(v.w);
  }
  __syncthreads();
#pragma unroll
  for (int i = 0; i < 2; i++) {
    int f = tid + i * 256;
    int cl = f >> 3, q4 = f & 7;
    uint4 v = *(const uint4*)(&tile[cl][q4 * 8]);
    *(uint4*)(vT + ((size_t)bh * 256 + c0 + cl) * Tc + t0 + q4 * 8) = v;
  }
}

// =====================================================================
// Kernel 3: flash attention. K/V tiles DMA'd to LDS by the TDM
// (tensor_load_to_lds + s_wait_tensorcnt), WMMA QK^T and P*V.
// =====================================================================
__global__ __launch_bounds__(128) void attn_kernel(
    const u16* __restrict__ qh, const u16* __restrict__ kh,
    const u16* __restrict__ vT, float* __restrict__ yb) {
  __shared__ alignas(16) u16 Qs[64 * 128];     // [qrow][hd] 16 KB
  __shared__ alignas(16) u16 Ks[32 * 128];     // [kv][hd]   8 KB
  __shared__ alignas(16) u16 VTs[256 * 32];    // [col][kv]  16 KB
  __shared__ alignas(16) u16 Ps[4][16 * 32];   // per-wave P 4 KB

  const int tid = threadIdx.x, lane = tid & 31, wid = tid >> 5;
  const int half = lane >> 4, l15 = lane & 15;
  const int qt = blockIdx.x, hp = blockIdx.y;
  const int b = blockIdx.z >> 1, br = blockIdx.z & 1;
  const int ko1 = 8 * half, ko2 = 16 + 8 * half;

  const size_t headOff = ((size_t)(b * 2 + br) * HPc + hp) * Tc * HD;
  const u16* qpb = qh + headOff;
  const u16* kpb = kh + headOff;
  const u16* vtb = vT + (size_t)(b * HPc + hp) * 256 * Tc;
  const int qrow0 = qt * 64 + wid * 16;
  const u32 ldsK = (u32)(size_t)(void*)Ks;
  const u32 ldsV = (u32)(size_t)(void*)VTs;

  // Q tile (64 x 128 bf16): straight b128 copy, once.
#pragma unroll
  for (int i = 0; i < 8; i++) {
    int f = tid + i * 128;              // 1024 uint4
    int row = f >> 4, q4 = f & 15;
    *(uint4*)(&Qs[row * 128 + q4 * 8]) =
        *(const uint4*)(qpb + (size_t)(qt * 64 + row) * HD + q4 * 8);
  }

  v8f o[16];
#pragma unroll
  for (int t = 0; t < 16; t++)
#pragma unroll
    for (int e = 0; e < 8; e++) o[t][e] = 0.f;
  float mrow[8], lrow[8];
#pragma unroll
  for (int e = 0; e < 8; e++) { mrow[e] = -3.0e38f; lrow[e] = 0.f; }

  const float scale = 0.08838834764831845f;  // 128^-0.5
  const int nck = (qt + 1) * 2;
  const u16* qrows = Qs + (wid * 16 + l15) * 128;

  for (int ck = 0; ck < nck; ck++) {
    const int kvb = ck * 32;
    __syncthreads();  // previous chunk fully consumed before TDM overwrites LDS
    if (wid == 0) {
      // K chunk: 32 rows x 128 bf16, contiguous rows -> Ks[kv][hd]
      tdm_load_2d_bf16(kpb + (size_t)kvb * HD, ldsK, /*x=*/HD, /*y=*/32, /*s0=*/HD);
      // V chunk: 256 cols x 32 bf16 (stride T between cols) -> VTs[col][kv]
      tdm_load_2d_bf16(vtb + kvb, ldsV, /*x=*/32, /*y=*/256, /*s0=*/Tc);
      __builtin_amdgcn_s_wait_tensorcnt(0);
    }
    __syncthreads();

    // S = Q * K^T
    v8f s0, s1;
#pragma unroll
    for (int e = 0; e < 8; e++) { s0[e] = 0.f; s1[e] = 0.f; }
#pragma unroll
    for (int s = 0; s < 4; s++) {
      v16bf qa = ldsFrag(qrows + s * 32, ko1, ko2);
      v16bf b0 = ldsFrag(Ks + ( 0 + l15) * 128 + s * 32, ko1, ko2);
      v16bf b1 = ldsFrag(Ks + (16 + l15) * 128 + s * 32, ko1, ko2);
      s0 = wmma_bf16(qa, b0, s0);
      s1 = wmma_bf16(qa, b1, s1);
    }

    // online softmax: DPP butterflies across 16-lane halves
    float alpha[8];
#pragma unroll
    for (int e = 0; e < 8; e++) {
      int row = qrow0 + e + 8 * half;
      int c0 = kvb + l15, c1 = kvb + 16 + l15;
      float v0 = (c0 <= row) ? s0[e] * scale : -3.0e38f;
      float v1 = (c1 <= row) ? s1[e] * scale : -3.0e38f;
      float rm = red16_max(fmaxf(v0, v1));
      float mn = fmaxf(mrow[e], rm);
      float al = __expf(mrow[e] - mn);
      float p0 = __expf(v0 - mn), p1 = __expf(v1 - mn);
      float rs = red16_sum(p0 + p1);
      lrow[e] = lrow[e] * al + rs;
      mrow[e] = mn;
      alpha[e] = al;
      int rl = e + 8 * half;
      Ps[wid][rl * 32 + l15]      = f2bf(p0);
      Ps[wid][rl * 32 + 16 + l15] = f2bf(p1);
    }
    __syncthreads();

    // O = O*alpha + P*V
    FragU pf;
    {
      const u16* pb = Ps[wid] + l15 * 32;
      pf.q[0] = *(const uint4*)(pb + ko1);
      pf.q[1] = *(const uint4*)(pb + ko2);
    }
#pragma unroll
    for (int t = 0; t < 16; t++) {
      v16bf vf = ldsFrag(VTs + (t * 16 + l15) * 32, ko1, ko2);
#pragma unroll
      for (int e = 0; e < 8; e++) o[t][e] *= alpha[e];
      o[t] = wmma_bf16(pf.v, vf, o[t]);
    }
  }

  float linv[8];
#pragma unroll
  for (int e = 0; e < 8; e++) linv[e] = fast_rcp(lrow[e]);
  const size_t yhead = ((size_t)(br * Bc + b) * HPc + hp) * Tc * 256;
#pragma unroll
  for (int t = 0; t < 16; t++)
#pragma unroll
    for (int e = 0; e < 8; e++) {
      int row = qrow0 + e + 8 * half;
      int col = t * 16 + l15;
      yb[yhead + (size_t)row * 256 + col] = o[t][e] * linv[e];
    }
}

// =====================================================================
// Kernel 4: y = rmsnorm((y1 - lam*y2) * silu_gate) * (1-LAMBDA_INIT)
// =====================================================================
__global__ __launch_bounds__(256) void epilogue_kernel(
    const float* __restrict__ yb, const float* __restrict__ g,
    const float* __restrict__ lq1, const float* __restrict__ lk1,
    const float* __restrict__ lq2, const float* __restrict__ lk2,
    float* __restrict__ out) {
  __shared__ float red[256];
  const int c = threadIdx.x;
  const int lin = blockIdx.x;
  const int hp = lin & 7;
  const int t = (lin >> 3) & (Tc - 1);
  const int b = lin >> 14;
  const size_t row = (size_t)(b * Tc + t);

  const size_t y1i = (((size_t)b * HPc + hp) * Tc + t) * 256 + c;
  const size_t y2off = (size_t)Bc * HPc * Tc * 256;
  const float y1 = yb[y1i];
  const float y2 = yb[y1i + y2off];
  const float gv = g[row * DM + hp * 256 + c];  // already SiLU'd

  float s1 = 0.f, s2 = 0.f;
#pragma unroll 8
  for (int j = 0; j < 64; j++) {
    s1 += lq1[hp * 64 + j] * lk1[hp * 64 + j];
    s2 += lq2[hp * 64 + j] * lk2[hp * 64 + j];
  }
  const float lam = __expf(s1) - __expf(s2) + LAMBDA_INIT;
  const float y = (y1 - lam * y2) * gv;

  red[c] = y * y;
  __syncthreads();
  for (int s = 128; s > 0; s >>= 1) {
    if (c < s) red[c] += red[c + s];
    __syncthreads();
  }
  const float rms = rsqrtf(red[0] * (1.f / 256.f) + EPSc);
  out[row * DM + hp * 256 + c] = y * rms * (1.f - LAMBDA_INIT);
}

// =====================================================================
extern "C" void kernel_launch(void* const* d_in, const int* in_sizes, int n_in,
                              void* d_out, int out_size, void* d_ws, size_t ws_size,
                              hipStream_t stream) {
  const float* x    = (const float*)d_in[0];
  const float* Wqkv = (const float*)d_in[1];
  const float* lq1  = (const float*)d_in[2];
  const float* lk1  = (const float*)d_in[3];
  const float* lq2  = (const float*)d_in[4];
  const float* lk2  = (const float*)d_in[5];
  const float* Wg   = (const float*)d_in[6];
  float* out = (float*)d_out;
  char* wsb  = (char*)d_ws;

  const int M = Bc * Tc;  // 4096
  size_t off = 0;
  float* qkv = (float*)(wsb + off); off += (size_t)M * QKVN * 4;
  float* gbf = (float*)(wsb + off); off += (size_t)M * DM * 4;
  u16*   qh  = (u16*)(wsb + off);   off += (size_t)Bc * 2 * HPc * Tc * HD * 2;
  u16*   kh  = (u16*)(wsb + off);   off += (size_t)Bc * 2 * HPc * Tc * HD * 2;
  u16*   vT  = (u16*)(wsb + off);   off += (size_t)Bc * HPc * 256 * Tc * 2;
  float* yb  = (float*)(wsb + off);

  gemm_bf16_kernel<0><<<dim3(QKVN / 128, M / 128), 256, 0, stream>>>(
      x, Wqkv, qkv, M, QKVN, DM);
  gemm_bf16_kernel<1><<<dim3(DM / 128, M / 128), 256, 0, stream>>>(
      x, Wg, gbf, M, DM, DM);
  prep_qk_kernel<<<Bc * Tc * NH, 128, 0, stream>>>(qkv, qh, kh);
  vtrans_kernel<<<dim3(Tc / 64, 4, Bc * HPc), 256, 0, stream>>>(qkv, vT);
  attn_kernel<<<dim3(Tc / 64, HPc, Bc * 2), 128, 0, stream>>>(qh, kh, vT, yb);
  epilogue_kernel<<<Bc * Tc * HPc, 256, 0, stream>>>(yb, gbf, lq1, lk1, lq2, lk2, out);
}
